// ListNetReranker_88021059764793
// MI455X (gfx1250) — matrix operand order repaired and checked by reading, not compiled, not run on hardware
//
#include <hip/hip_runtime.h>
#include <hip/hip_bf16.h>
#include <math.h>

// ---------------------------------------------------------------------------
// ListNet reranker loss on gfx1250: TDM-staged bf16 WMMA MLP scorer +
// deterministic group-loss reductions.
// ---------------------------------------------------------------------------

typedef __attribute__((ext_vector_type(16))) __bf16 v16bf;
typedef __attribute__((ext_vector_type(8)))  float  v8f;
typedef __attribute__((ext_vector_type(4)))  unsigned int v4u;
typedef __attribute__((ext_vector_type(8)))  int    v8i;
typedef __attribute__((ext_vector_type(4)))  int    v4i;

#define N_TOT   2097152
#define K_GRP   16
#define G_GRP   (N_TOT / K_GRP)       // 131072 groups
#define D_IN    32
#define HID     96
#define M_TOP   4
#define K_NEG   16
#define EPS_F   1e-9f

#define WAVES_PER_BLOCK 4             // 128-thread blocks

// LDS row strides (elements), padded for bank spread.
constexpr int W1T_STRIDE = 40;   // W1^T: [96 rows][32 cols] bf16
constexpr int W2T_STRIDE = 104;  // W2^T: [96 rows][96 cols] bf16
constexpr int STG_STRIDE = 104;  // per-wave activation staging: [16][96] bf16
constexpr int XSTG_STRIDE = 33;  // per-wave X staging: [16][32] f32, TDM-padded

// A-operand (16-bit A 16x32), contiguous K order matching B:
//  lanes 0-15 : elems 0..15 -> K 0..15   (kbaseA=0)
//  lanes 16-31: elems 0..15 -> K 16..31  (kbaseA=16)
__device__ __forceinline__ v16bf load_A_tile(const __bf16* __restrict__ row,
                                             int kchunk, int kbaseA) {
  v16bf a;
#pragma unroll
  for (int i = 0; i < 8; ++i) {
    int k = kchunk + kbaseA + 2 * i;
    a[2 * i]     = row[k];
    a[2 * i + 1] = row[k + 1];
  }
  return a;
}

// Layer-1 A operand straight from TDM-staged f32 X tile (pairwise cvt).
__device__ __forceinline__ v16bf load_A_from_f32(const float* __restrict__ row,
                                                 int kbaseA) {
  v16bf a;
#pragma unroll
  for (int i = 0; i < 8; ++i) {
    int k = kbaseA + 2 * i;
    a[2 * i]     = (__bf16)row[k];
    a[2 * i + 1] = (__bf16)row[k + 1];
  }
  return a;
}

// B-operand: lane holds column n = lane&15; lanes 0-15 K=e, lanes 16-31 K=16+e.
__device__ __forceinline__ v16bf load_B_tile(const __bf16* __restrict__ p) {
  v16bf b;
#pragma unroll
  for (int i = 0; i < 16; ++i) b[i] = p[i];
  return b;
}

// SiLU with v_rcp_f32 instead of the full IEEE divide sequence.
__device__ __forceinline__ float silu_f(float x) {
  return x * __builtin_amdgcn_rcpf(1.0f + __expf(-x));
}

// ---------------------------------------------------------------------------
// Kernel 1: MLP scorer. One wave per 16-row tile. X tiles DMA'd by the
// Tensor Data Mover into padded LDS; bf16 WMMA with f32 accumulate.
// ---------------------------------------------------------------------------
__global__ __launch_bounds__(128) void mlp_score_kernel(
    const float* __restrict__ X,
    const float* __restrict__ W1, const float* __restrict__ b1,
    const float* __restrict__ W2, const float* __restrict__ b2,
    const float* __restrict__ W3, const float* __restrict__ b3,
    float* __restrict__ s_out) {
  __shared__ __bf16 w1t[HID * W1T_STRIDE];
  __shared__ __bf16 w2t[HID * W2T_STRIDE];
  __shared__ float  w3s[HID], b1s[HID], b2s[HID];
  __shared__ float  b3s;
  __shared__ __bf16 stage[WAVES_PER_BLOCK][16 * STG_STRIDE];
  __shared__ float  xstg[WAVES_PER_BLOCK][16 * XSTG_STRIDE];

  const int tid = threadIdx.x;

  // Stage weights transposed (bf16) + biases (f32) into LDS.
  for (int i = tid; i < D_IN * HID; i += 128) {
    int k = i / HID, h = i - k * HID;               // W1[k][h]
    w1t[h * W1T_STRIDE + k] = (__bf16)W1[i];
  }
  for (int i = tid; i < HID * HID; i += 128) {
    int k = i / HID, h = i - k * HID;               // W2[k][h]
    w2t[h * W2T_STRIDE + k] = (__bf16)W2[i];
  }
  for (int i = tid; i < HID; i += 128) {
    w3s[i] = W3[i];
    b1s[i] = b1[i];
    b2s[i] = b2[i];
  }
  if (tid == 0) b3s = b3[0];
  __syncthreads();

  const int lane   = tid & 31;
  const int wave   = tid >> 5;
  const int m      = lane & 15;       // row (A) or column (B/C) index
  const int hiHalf = lane >> 4;       // 0: low half-wave, 1: high half-wave
  const int kbaseA = hiHalf * 16;
  const int kbaseB = hiHalf * 16;
  __bf16* stg = stage[wave];
  const __bf16* arow = &stage[wave][m * STG_STRIDE];
  const float*  xrow = &xstg[wave][m * XSTG_STRIDE];
  const unsigned xstg_lds_addr = (unsigned)(size_t)(void*)&xstg[wave][0];

  const int gwave  = blockIdx.x * WAVES_PER_BLOCK + wave;
  const int nwaves = gridDim.x * WAVES_PER_BLOCK;
  const int NT     = N_TOT / 16;

  // TDM D# group1 (constant per wave): data_size=4B, pad every 32 DWORDs by
  // 1 DWORD (row stride 33 f32 -> conflict-free banks), tensor 32x16,
  // tile 32x16, dim0 stride 32 elements.
  v8i tg1;
  tg1[0] = (2 << 16)            // data_size = 4 bytes
         | (1 << 20)            // pad_enable
         | (4 << 22)            // pad_interval: 32 DWORDs (128 B) per row
         | (0 << 25);           // pad_amount: 1 DWORD
  tg1[1] = (D_IN & 0xFFFF) << 16;            // tensor_dim0[15:0] = 32
  tg1[2] = (16 << 16);                       // tensor_dim0 hi=0 | tensor_dim1=16
  tg1[3] = (D_IN << 16);                     // tensor_dim1 hi=0 | tile_dim0=32
  tg1[4] = 16;                               // tile_dim1=16 | tile_dim2=0
  tg1[5] = D_IN;                             // tensor_dim0_stride = 32
  tg1[6] = 0;
  tg1[7] = 0;
  const v4i tzero4 = {0, 0, 0, 0};
  const v8i tzero8 = {0, 0, 0, 0, 0, 0, 0, 0};

  for (int tile = gwave; tile < NT; tile += nwaves) {
    const int r0 = tile * 16;

    // ---- TDM: DMA the 16x32 f32 X tile into padded LDS ----
    const unsigned long long gaddr =
        (unsigned long long)(size_t)(X + (size_t)r0 * D_IN);
    v4u tg0;
    tg0[0] = 1u;                                   // count=1 valid descriptor
    tg0[1] = xstg_lds_addr;                        // lds_addr
    tg0[2] = (unsigned)(gaddr & 0xFFFFFFFFull);    // global_addr[31:0]
    tg0[3] = (unsigned)((gaddr >> 32) & 0x1FFFFFFull) | (2u << 30); // type=2
    __builtin_amdgcn_tensor_load_to_lds(tg0, tg1, tzero4, tzero4, tzero8, 0);
    __builtin_amdgcn_s_wait_tensorcnt(0);

    v16bf a1 = load_A_from_f32(xrow, kbaseA);

    // ---- Layer 1: (16x32) @ (32x96) + b1, SiLU -> stage ----
#pragma unroll
    for (int j = 0; j < 6; ++j) {
      v8f acc;
      float bj = b1s[16 * j + m];
#pragma unroll
      for (int v = 0; v < 8; ++v) acc[v] = bj;     // bias folded into C
      v16bf bt = load_B_tile(&w1t[(16 * j + m) * W1T_STRIDE + kbaseB]);
      acc = __builtin_amdgcn_wmma_f32_16x16x32_bf16(
          false, a1, false, bt, (short)0, acc, false, false);
#pragma unroll
      for (int v = 0; v < 8; ++v) {
        float h = silu_f(acc[v]);
        stg[(v + 8 * hiHalf) * STG_STRIDE + 16 * j + m] = (__bf16)h;
      }
    }

    // Re-load h1 in A layout (3 K-chunks of 32).
    v16bf a2_0 = load_A_tile(arow, 0,  kbaseA);
    v16bf a2_1 = load_A_tile(arow, 32, kbaseA);
    v16bf a2_2 = load_A_tile(arow, 64, kbaseA);

    // ---- Layer 2: (16x96) @ (96x96) + b2, SiLU -> stage ----
#pragma unroll
    for (int j = 0; j < 6; ++j) {
      v8f acc;
      float bj = b2s[16 * j + m];
#pragma unroll
      for (int v = 0; v < 8; ++v) acc[v] = bj;
      const __bf16* wb = &w2t[(16 * j + m) * W2T_STRIDE];
      acc = __builtin_amdgcn_wmma_f32_16x16x32_bf16(
          false, a2_0, false, load_B_tile(wb + kbaseB), (short)0, acc, false, false);
      acc = __builtin_amdgcn_wmma_f32_16x16x32_bf16(
          false, a2_1, false, load_B_tile(wb + 32 + kbaseB), (short)0, acc, false, false);
      acc = __builtin_amdgcn_wmma_f32_16x16x32_bf16(
          false, a2_2, false, load_B_tile(wb + 64 + kbaseB), (short)0, acc, false, false);
#pragma unroll
      for (int v = 0; v < 8; ++v) {
        float h = silu_f(acc[v]);
        stg[(v + 8 * hiHalf) * STG_STRIDE + 16 * j + m] = (__bf16)h;
      }
    }

    // ---- Layer 3: (16x96) @ (96x1) + b3 (VALU dot, half-wave split) ----
    float partial = 0.0f;
    const int k0 = 48 * hiHalf;
#pragma unroll
    for (int k = 0; k < 48; ++k)
      partial += (float)arow[k0 + k] * w3s[k0 + k];
    float total = partial + __shfl_xor(partial, 16, 32);
    if (hiHalf == 0) s_out[r0 + m] = total + b3s;
  }
}

// ---------------------------------------------------------------------------
// Kernel 2: per-group ListNet CE + sampled pairwise loss. One thread / group.
// Deterministic: per-block tree reduction, partials written (no atomics).
// ---------------------------------------------------------------------------
__global__ __launch_bounds__(256) void group_loss_kernel(
    const float* __restrict__ s, const float* __restrict__ y,
    const int* __restrict__ rand_idx, float* __restrict__ partials) {
  __shared__ float ssh[256 * 17];
  __shared__ float tsh[256 * 17];
  __shared__ float r0[256], r1[256], r2[256];

  const int tid = threadIdx.x;
  const int g = blockIdx.x * 256 + tid;

  float ce = 0.0f, pair = 0.0f, cnt = 0.0f;

  if (g < G_GRP) {
    float* sv = &ssh[tid * 17];
    float* tv = &tsh[tid * 17];
    const size_t base = (size_t)g * K_GRP;
#pragma unroll
    for (int i = 0; i < K_GRP; ++i) {
      sv[i] = s[base + i];
      tv[i] = y[base + i];
    }

    // Raw-exp softmax sums (faithful to reference: no max-subtract).
    float den_s = 0.0f, den_t = 0.0f;
#pragma unroll
    for (int i = 0; i < K_GRP; ++i) {
      den_s += __expf(sv[i]);
      den_t += __expf(tv[i]);
    }
    const float rs = __builtin_amdgcn_rcpf(den_s + EPS_F);
    const float rt = __builtin_amdgcn_rcpf(den_t + EPS_F);
#pragma unroll
    for (int i = 0; i < K_GRP; ++i) {
      float p = __expf(sv[i]) * rs;
      float q = __expf(tv[i]) * rt;
      ce += -q * __logf(p + EPS_F);
    }

    // Top-4 targets; ties -> lowest index (matches lax.top_k).
    unsigned topmask = 0u;
    int topidx[M_TOP];
#pragma unroll
    for (int mm = 0; mm < M_TOP; ++mm) {
      float best = -3.4e38f;
      int bi = 0;
#pragma unroll
      for (int i = 0; i < K_GRP; ++i) {
        bool taken = (topmask >> i) & 1u;
        if (!taken && tv[i] > best) { best = tv[i]; bi = i; }
      }
      topmask |= 1u << bi;
      topidx[mm] = bi;
    }

    // Sampled negatives: stable softplus over non-top samples.
    const int* ri = rand_idx + (size_t)g * (M_TOP * K_NEG);
#pragma unroll
    for (int mm = 0; mm < M_TOP; ++mm) {
      float pos = sv[topidx[mm]];
#pragma unroll
      for (int k = 0; k < K_NEG; ++k) {
        int r = ri[mm * K_NEG + k] & 15;
        if (!((topmask >> r) & 1u)) {
          float nd = sv[r] - pos;                       // -diff
          float sp = fmaxf(nd, 0.0f) + log1pf(__expf(-fabsf(nd)));
          pair += sp;
          cnt  += 1.0f;
        }
      }
    }
  }

  r0[tid] = ce; r1[tid] = pair; r2[tid] = cnt;
  __syncthreads();
  for (int off = 128; off > 0; off >>= 1) {
    if (tid < off) {
      r0[tid] += r0[tid + off];
      r1[tid] += r1[tid + off];
      r2[tid] += r2[tid + off];
    }
    __syncthreads();
  }
  if (tid == 0) {
    partials[blockIdx.x * 3 + 0] = r0[0];
    partials[blockIdx.x * 3 + 1] = r1[0];
    partials[blockIdx.x * 3 + 2] = r2[0];
  }
}

// ---------------------------------------------------------------------------
// Kernel 3: fixed-order final reduction + scalar combine.
// ---------------------------------------------------------------------------
__global__ __launch_bounds__(256) void finalize_kernel(
    const float* __restrict__ partials, int nblocks, float* __restrict__ out) {
  __shared__ float r0[256], r1[256], r2[256];
  const int tid = threadIdx.x;
  float a = 0.0f, b = 0.0f, c = 0.0f;
  for (int i = tid; i < nblocks; i += 256) {
    a += partials[i * 3 + 0];
    b += partials[i * 3 + 1];
    c += partials[i * 3 + 2];
  }
  r0[tid] = a; r1[tid] = b; r2[tid] = c;
  __syncthreads();
  for (int off = 128; off > 0; off >>= 1) {
    if (tid < off) {
      r0[tid] += r0[tid + off];
      r1[tid] += r1[tid + off];
      r2[tid] += r2[tid + off];
    }
    __syncthreads();
  }
  if (tid == 0) {
    float denom = fmaxf(r2[0], 1.0f);
    out[0] = 1.0f * (r0[0] / (float)G_GRP) + 0.5f * (r1[0] / denom);
  }
}

// ---------------------------------------------------------------------------
extern "C" void kernel_launch(void* const* d_in, const int* in_sizes, int n_in,
                              void* d_out, int out_size, void* d_ws, size_t ws_size,
                              hipStream_t stream) {
  (void)in_sizes; (void)n_in; (void)out_size; (void)ws_size;
  const float* X  = (const float*)d_in[0];
  const float* y  = (const float*)d_in[1];
  const float* W1 = (const float*)d_in[2];
  const float* b1 = (const float*)d_in[3];
  const float* W2 = (const float*)d_in[4];
  const float* b2 = (const float*)d_in[5];
  const float* W3 = (const float*)d_in[6];
  const float* b3 = (const float*)d_in[7];
  // d_in[8] = group (int64): unused, groups are contiguous blocks of 16.
  const int* rand_idx = (const int*)d_in[9];
  float* out = (float*)d_out;

  float* s_buf    = (float*)d_ws;        // N_TOT floats (8 MB)
  float* partials = s_buf + N_TOT;       // 512 * 3 floats
  const int nblocks2 = G_GRP / 256;      // 512

  mlp_score_kernel<<<4096, 128, 0, stream>>>(X, W1, b1, W2, b2, W3, b3, s_buf);
  group_loss_kernel<<<nblocks2, 256, 0, stream>>>(s_buf, y, rand_idx, partials);
  finalize_kernel<<<1, 256, 0, stream>>>(partials, nblocks2, out);
}
